// MultiHeadAttention_44324062495235
// MI455X (gfx1250) — matrix operand order repaired and checked by reading, not compiled, not run on hardware
//
#include <hip/hip_runtime.h>
#include <hip/hip_bf16.h>

// ---------------------------------------------------------------------------
// MHA forward for MI455X (gfx1250, wave32, WMMA + TDM).
// Pipeline:
//   1) Q/K/V projections:  bf16 WMMA GEMM (fp32 -> bf16 convert in LDS stage)
//   2) flash-attention:    online softmax, Q.K^T and P.V via bf16 WMMA,
//                          K/V tiles staged by the Tensor Data Mover
//   3) output projection:  bf16 WMMA GEMM, fp32 out
// Workspace: 4 x [4096,1024] bf16 buffers = 33,554,432 bytes.
// ---------------------------------------------------------------------------

typedef __bf16 bf16;
typedef bf16  v16bf  __attribute__((ext_vector_type(16)));
typedef bf16  bf16x8 __attribute__((ext_vector_type(8)));
typedef bf16  bf16x4 __attribute__((ext_vector_type(4)));
typedef float v8f    __attribute__((ext_vector_type(8)));
typedef float f32x4  __attribute__((ext_vector_type(4)));
typedef unsigned int u32x4 __attribute__((ext_vector_type(4)));
typedef int   i32x8  __attribute__((ext_vector_type(8)));
typedef int   i32x4  __attribute__((ext_vector_type(4)));

#define D_MODEL 1024
#define N_HEADS 16
#define D_HEAD  64
#define SEQ     2048
#define BATCH   2

#if defined(__AMDGCN__) && __has_builtin(__builtin_amdgcn_tensor_load_to_lds)
#define USE_TDM 1
#else
#define USE_TDM 0
#endif

__device__ __forceinline__ v8f wmma_bf16(v16bf a, v16bf b, v8f c) {
  // D = A(16x32 bf16) * B(32x16 bf16) + C(16x16 f32)
  return __builtin_amdgcn_wmma_f32_16x16x32_bf16(false, a, false, b,
                                                 (short)0, c, false, false);
}

// A-fragment (16x32, MxK): lane L holds row M=L%16; with h=L/16 the lane's
// 16 elements are K = 8h+{0..7} then K = 8h+16+{0..7}.  `row` points at K=0.
__device__ __forceinline__ v16bf load_afrag(const bf16* row, int half) {
  bf16x8 lo = *(const bf16x8*)(row + 8 * half);
  bf16x8 hi = *(const bf16x8*)(row + 8 * half + 16);
  v16bf r;
#pragma unroll
  for (int i = 0; i < 8; ++i) { r[i] = lo[i]; r[i + 8] = hi[i]; }
  return r;
}

// B-fragment (32x16, KxN): lane L holds col N=L%16; elements K = 8h+{0..7} then K = 8h+16+{0..7}.
__device__ __forceinline__ v16bf load_bfrag(const bf16* row, int half) {
  bf16x8 lo = *(const bf16x8*)(row + 8 * half);
  bf16x8 hi = *(const bf16x8*)(row + 8 * half + 16);
  v16bf r;
#pragma unroll
  for (int i = 0; i < 8; ++i) { r[i] = lo[i]; r[i + 8] = hi[i]; }
  return r;
}

// ---------------------------------------------------------------------------
// C[M,N] = A[M,K] @ B[N,K]^T + bias   (torch Linear, NT GEMM).
// Block tile 128x128, BK=32, 8 waves, each wave 64x32 (4x2 WMMA tiles).
// ---------------------------------------------------------------------------
template <bool A_BF16, bool OUT_BF16>
__global__ __launch_bounds__(256)
void gemm_nt_bias(const void* __restrict__ Ap, const float* __restrict__ Bw,
                  const float* __restrict__ bias, void* __restrict__ Cp,
                  int M, int N, int K) {
  __shared__ bf16 sA[128][40];   // +8 pad: 16 rows hit distinct banks
  __shared__ bf16 sB[128][40];

  const int tid   = threadIdx.x;
  const int lane  = tid & 31;
  const int wv    = tid >> 5;          // 0..7
  const int laneN = lane & 15;
  const int half  = lane >> 4;
  const int bm = blockIdx.y * 128, bn = blockIdx.x * 128;
  const int wm = (wv >> 2) * 64, wn = (wv & 3) * 32;

  v8f acc[4][2];
#pragma unroll
  for (int mt = 0; mt < 4; ++mt)
#pragma unroll
    for (int nt = 0; nt < 2; ++nt)
#pragma unroll
      for (int i = 0; i < 8; ++i) acc[mt][nt][i] = 0.f;

  for (int kt = 0; kt < K; kt += 32) {
    __syncthreads();
    // ---- stage A tile (convert to bf16 if fp32) ----
    if (A_BF16) {
      const bf16* A = (const bf16*)Ap;
#pragma unroll
      for (int i = 0; i < 2; ++i) {
        int c = tid + i * 256; int r = c >> 2; int cc = (c & 3) * 8;
        *(bf16x8*)&sA[r][cc] =
            *(const bf16x8*)(A + (size_t)(bm + r) * K + kt + cc);
      }
    } else {
      const float* A = (const float*)Ap;
#pragma unroll
      for (int i = 0; i < 4; ++i) {
        int c = tid + i * 256; int r = c >> 3; int cc = (c & 7) * 4;
        f32x4 f = *(const f32x4*)(A + (size_t)(bm + r) * K + kt + cc);
        bf16x4 hh; hh[0] = (bf16)f[0]; hh[1] = (bf16)f[1];
        hh[2] = (bf16)f[2]; hh[3] = (bf16)f[3];
        *(bf16x4*)&sA[r][cc] = hh;
      }
    }
    // ---- stage B (weight) tile, fp32 -> bf16 ----
#pragma unroll
    for (int i = 0; i < 4; ++i) {
      int c = tid + i * 256; int r = c >> 3; int cc = (c & 7) * 4;
      f32x4 f = *(const f32x4*)(Bw + (size_t)(bn + r) * K + kt + cc);
      bf16x4 hh; hh[0] = (bf16)f[0]; hh[1] = (bf16)f[1];
      hh[2] = (bf16)f[2]; hh[3] = (bf16)f[3];
      *(bf16x4*)&sB[r][cc] = hh;
    }
    // ---- prefetch next k-tile (global_prefetch_b8) ----
    if (kt + 32 < K) {
      if (!A_BF16)
        __builtin_prefetch((const float*)Ap + (size_t)(bm + (tid >> 1)) * K +
                               kt + 32 + (tid & 1) * 16, 0, 1);
      __builtin_prefetch(Bw + (size_t)(bn + (tid >> 1)) * K + kt + 32 +
                             (tid & 1) * 16, 0, 1);
    }
    __syncthreads();

    v16bf af[4], bfr[2];
#pragma unroll
    for (int mt = 0; mt < 4; ++mt)
      af[mt] = load_afrag(&sA[wm + mt * 16 + laneN][0], half);
#pragma unroll
    for (int nt = 0; nt < 2; ++nt)
      bfr[nt] = load_bfrag(&sB[wn + nt * 16 + laneN][0], half);
#pragma unroll
    for (int mt = 0; mt < 4; ++mt)
#pragma unroll
      for (int nt = 0; nt < 2; ++nt)
        acc[mt][nt] = wmma_bf16(af[mt], bfr[nt], acc[mt][nt]);
  }

  // ---- epilogue: bias + store (C layout: lane N=L%16, VGPR v -> M=v+8*half)
#pragma unroll
  for (int nt = 0; nt < 2; ++nt) {
    const int col = bn + wn + nt * 16 + laneN;
    const float bc = bias[col];
#pragma unroll
    for (int mt = 0; mt < 4; ++mt) {
#pragma unroll
      for (int v = 0; v < 8; ++v) {
        const int row = bm + wm + mt * 16 + v + 8 * half;
        const float o = acc[mt][nt][v] + bc;
        if (OUT_BF16) ((bf16*)Cp)[(size_t)row * N + col] = (bf16)o;
        else          ((float*)Cp)[(size_t)row * N + col] = o;
      }
    }
  }
}

// ---------------------------------------------------------------------------
// Flash attention forward. Grid: (SEQ/128, BATCH*N_HEADS). Block: 256 (8 waves).
// Each wave owns 16 query rows; kv streamed in tiles of 64 keys.
// K/V tiles are staged into LDS by the Tensor Data Mover (one 2-D D# per
// tile, LDS row padding 32+4 DWORDs done by the TDM itself).
// ---------------------------------------------------------------------------
__global__ __launch_bounds__(256)
void flash_attn_fwd(const bf16* __restrict__ Qp, const bf16* __restrict__ Kp,
                    const bf16* __restrict__ Vp, bf16* __restrict__ Op) {
  __shared__ bf16 sK [64][72];       // [key][dh]   (72 pad: conflict-free)
  __shared__ bf16 sV [64][72];       // [key][dh]   raw V tile (TDM dest)
  __shared__ bf16 sVT[64][72];       // [dh][key]
  __shared__ bf16 sP [8][16][72];    // per-wave P bounce buffer [row][key]

  const int tid   = threadIdx.x;
  const int lane  = tid & 31;
  const int wv    = tid >> 5;
  const int laneN = lane & 15;
  const int half  = lane >> 4;
  const int qblk  = blockIdx.x;
  const int b = blockIdx.y >> 4, h = blockIdx.y & 15;

  const size_t bhbase = (size_t)b * SEQ * D_MODEL + (size_t)h * D_HEAD;

  // Q A-fragments straight from global (rows are dh-contiguous).
  const int qrow = qblk * 128 + wv * 16 + laneN;
  const bf16* qptr = Qp + bhbase + (size_t)qrow * D_MODEL;
  v16bf qf[2];
#pragma unroll
  for (int s = 0; s < 2; ++s) qf[s] = load_afrag(qptr + s * 32, half);

  float m_run[8], l_run[8];
  v8f acco[4];
#pragma unroll
  for (int v = 0; v < 8; ++v) { m_run[v] = -3.0e38f; l_run[v] = 0.f; }
#pragma unroll
  for (int nt = 0; nt < 4; ++nt)
#pragma unroll
    for (int i = 0; i < 8; ++i) acco[nt][i] = 0.f;

  for (int kt = 0; kt < SEQ / 64; ++kt) {
    __syncthreads();
#if USE_TDM
    // ---- TDM: DMA K tile -> sK and V tile -> sV (wave 0 issues both) ----
    if (wv == 0) {
      // D# group1: workgroup_mask=0, data_size=2B, pad_enable,
      // pad_interval=32 DWORDs (enc 4), pad_amount=4 DWORDs (enc 3),
      // tensor_dim0=64, tensor_dim1=64, tile_dim0=64, tile_dim1=64,
      // tensor_dim0_stride = D_MODEL elements.
      i32x8 g1;
      g1[0] = (1 << 16) | (1 << 20) | (4 << 22) | (3 << 25);
      g1[1] = (64 << 16);            // tensor_dim0[15:0]
      g1[2] = (64 << 16);            // tensor_dim1[15:0]
      g1[3] = (64 << 16);            // tile_dim0
      g1[4] = 64;                    // tile_dim1 (tile_dim2 = 0)
      g1[5] = D_MODEL;               // tensor_dim0_stride[31:0]
      g1[6] = 0; g1[7] = 0;
      i32x4 gz;
      gz[0] = 0; gz[1] = 0; gz[2] = 0; gz[3] = 0;
      i32x8 gz8;
#pragma unroll
      for (int i = 0; i < 8; ++i) gz8[i] = 0;

      unsigned long long gaK =
          (unsigned long long)(Kp + bhbase + (size_t)(kt * 64) * D_MODEL);
      u32x4 g0;
      g0[0] = 1u;                                        // count=1
      g0[1] = (unsigned int)(unsigned long long)&sK[0][0];
      g0[2] = (unsigned int)gaK;
      g0[3] = (unsigned int)((gaK >> 32) & 0x1FFFFFFu) | (2u << 30);
      __builtin_amdgcn_tensor_load_to_lds(g0, g1, gz, gz, gz8, 0);

      unsigned long long gaV =
          (unsigned long long)(Vp + bhbase + (size_t)(kt * 64) * D_MODEL);
      g0[1] = (unsigned int)(unsigned long long)&sV[0][0];
      g0[2] = (unsigned int)gaV;
      g0[3] = (unsigned int)((gaV >> 32) & 0x1FFFFFFu) | (2u << 30);
      __builtin_amdgcn_tensor_load_to_lds(g0, g1, gz, gz, gz8, 0);

      __builtin_amdgcn_s_wait_tensorcnt(0);
    }
    __syncthreads();
    // ---- transpose sV -> sVT in LDS ----
#pragma unroll
    for (int i = 0; i < 2; ++i) {
      int c = tid + i * 256; int dh = c >> 3; int k0 = (c & 7) * 8;
      bf16x8 t;
#pragma unroll
      for (int j = 0; j < 8; ++j) t[j] = sV[k0 + j][dh];
      *(bf16x8*)&sVT[dh][k0] = t;
    }
    __syncthreads();
#else
    // ---- fallback staging: vector copy K, strided-gather V^T ----
#pragma unroll
    for (int i = 0; i < 2; ++i) {
      int c = tid + i * 256; int r = c >> 3; int cc = (c & 7) * 8;
      *(bf16x8*)&sK[r][cc] =
          *(const bf16x8*)(Kp + bhbase + (size_t)(kt * 64 + r) * D_MODEL + cc);
    }
#pragma unroll
    for (int i = 0; i < 2; ++i) {
      int c = tid + i * 256; int dh = c >> 3; int k0 = (c & 7) * 8;
      bf16x8 t;
#pragma unroll
      for (int j = 0; j < 8; ++j)
        t[j] = Vp[bhbase + (size_t)(kt * 64 + k0 + j) * D_MODEL + dh];
      *(bf16x8*)&sVT[dh][k0] = t;
    }
    __syncthreads();
#endif

    // ---- scores = Q @ K^T : 4 n-tiles x 2 k-steps ----
    v8f accs[4];
#pragma unroll
    for (int nt = 0; nt < 4; ++nt)
#pragma unroll
      for (int i = 0; i < 8; ++i) accs[nt][i] = 0.f;
#pragma unroll
    for (int s = 0; s < 2; ++s)
#pragma unroll
      for (int nt = 0; nt < 4; ++nt) {
        v16bf kf = load_bfrag(&sK[nt * 16 + laneN][s * 32], half);
        accs[nt] = wmma_bf16(qf[s], kf, accs[nt]);
      }

    // ---- scale + online softmax ----
#pragma unroll
    for (int nt = 0; nt < 4; ++nt)
#pragma unroll
      for (int v = 0; v < 8; ++v) accs[nt][v] *= 0.125f;

    float mnew[8], alpha[8], rsum[8];
#pragma unroll
    for (int v = 0; v < 8; ++v) {
      float mx = fmaxf(fmaxf(accs[0][v], accs[1][v]),
                       fmaxf(accs[2][v], accs[3][v]));
#pragma unroll
      for (int off = 8; off >= 1; off >>= 1)
        mx = fmaxf(mx, __shfl_xor(mx, off, 32));   // stays in 16-lane half
      mnew[v]  = fmaxf(m_run[v], mx);
      alpha[v] = __expf(m_run[v] - mnew[v]);
      m_run[v] = mnew[v];
      rsum[v]  = 0.f;
    }

    // P = exp(S-m); bounce through per-wave LDS to reach A-fragment layout.
#pragma unroll
    for (int nt = 0; nt < 4; ++nt)
#pragma unroll
      for (int v = 0; v < 8; ++v) {
        float p = __expf(accs[nt][v] - mnew[v]);
        rsum[v] += p;
        sP[wv][v + 8 * half][nt * 16 + laneN] = (bf16)p;
      }
#pragma unroll
    for (int v = 0; v < 8; ++v) {
      float t = rsum[v];
#pragma unroll
      for (int off = 8; off >= 1; off >>= 1) t += __shfl_xor(t, off, 32);
      l_run[v] = l_run[v] * alpha[v] + t;
    }
#pragma unroll
    for (int nt = 0; nt < 4; ++nt)
#pragma unroll
      for (int v = 0; v < 8; ++v) acco[nt][v] *= alpha[v];

    // ---- O += P @ V  (per-wave sP: DS in-order, no block barrier needed) ----
#pragma unroll
    for (int s = 0; s < 2; ++s) {
      v16bf pf = load_afrag(&sP[wv][laneN][s * 32], half);
#pragma unroll
      for (int nt = 0; nt < 4; ++nt) {
        v16bf vf = load_bfrag(&sVT[nt * 16 + laneN][s * 32], half);
        acco[nt] = wmma_bf16(pf, vf, acco[nt]);
      }
    }
  }

  // ---- normalize + store bf16 into [B,S,D_MODEL] (heads concatenated) ----
#pragma unroll
  for (int v = 0; v < 8; ++v) l_run[v] = 1.0f / l_run[v];
#pragma unroll
  for (int nt = 0; nt < 4; ++nt)
#pragma unroll
    for (int v = 0; v < 8; ++v) {
      const int row = qblk * 128 + wv * 16 + v + 8 * half;
      Op[bhbase + (size_t)row * D_MODEL + nt * 16 + laneN] =
          (bf16)(acco[nt][v] * l_run[v]);
    }
}

// ---------------------------------------------------------------------------
extern "C" void kernel_launch(void* const* d_in, const int* in_sizes, int n_in,
                              void* d_out, int out_size, void* d_ws,
                              size_t ws_size, hipStream_t stream) {
  const float* query = (const float*)d_in[0];
  const float* key_  = (const float*)d_in[1];
  const float* value = (const float*)d_in[2];
  const float* Wq = (const float*)d_in[3]; const float* bq = (const float*)d_in[4];
  const float* Wk = (const float*)d_in[5]; const float* bk = (const float*)d_in[6];
  const float* Wv = (const float*)d_in[7]; const float* bv = (const float*)d_in[8];
  const float* Wo = (const float*)d_in[9]; const float* bo = (const float*)d_in[10];

  const int M = BATCH * SEQ;            // 4096
  const size_t nelem = (size_t)M * D_MODEL;
  bf16* Qp = (bf16*)d_ws;               // 4 bf16 buffers: 4*8 MiB = 32 MiB
  bf16* Kp = Qp + nelem;
  bf16* Vp = Kp + nelem;
  bf16* Ao = Vp + nelem;

  dim3 blk(256);
  dim3 gg(D_MODEL / 128, M / 128);      // (8, 32)

  gemm_nt_bias<false, true><<<gg, blk, 0, stream>>>(query, Wq, bq, Qp,
                                                    M, D_MODEL, D_MODEL);
  gemm_nt_bias<false, true><<<gg, blk, 0, stream>>>(key_,  Wk, bk, Kp,
                                                    M, D_MODEL, D_MODEL);
  gemm_nt_bias<false, true><<<gg, blk, 0, stream>>>(value, Wv, bv, Vp,
                                                    M, D_MODEL, D_MODEL);

  flash_attn_fwd<<<dim3(SEQ / 128, BATCH * N_HEADS), blk, 0, stream>>>(
      Qp, Kp, Vp, Ao);

  gemm_nt_bias<true, false><<<gg, blk, 0, stream>>>(Ao, Wo, bo, d_out,
                                                    M, D_MODEL, D_MODEL);
}
